// GNNActorVariablePrice_57861799412273
// MI455X (gfx1250) — matrix operand order repaired and checked by reading, not compiled, not run on hardware
//
#include <hip/hip_runtime.h>
#include <hip/hip_bf16.h>
#include <stdint.h>

// ---- problem sizes (match reference) ----
#define NREG 8192
#define CIN  128
#define CMID 256
#define PM   8

// ---- types for WMMA / TDM ----
typedef __bf16 bf16_t;
typedef bf16_t bf16x8  __attribute__((ext_vector_type(8)));
typedef bf16_t bf16x16 __attribute__((ext_vector_type(16)));
typedef float  f32x8   __attribute__((ext_vector_type(8)));
typedef float  f32x2   __attribute__((ext_vector_type(2)));
typedef unsigned int u32x4 __attribute__((ext_vector_type(4)));
typedef int    i32x4   __attribute__((ext_vector_type(4)));
typedef int    i32x8   __attribute__((ext_vector_type(8)));

union BF16V { bf16x16 v; bf16x8 h[2]; };

__device__ __forceinline__ bf16_t to_bf(float f) { return (bf16_t)f; }

// ------------------------------------------------------------------
// degree / normalization
// ------------------------------------------------------------------
__global__ void k_deg_init(float* deg) {
    int i = blockIdx.x * blockDim.x + threadIdx.x;
    if (i < NREG) deg[i] = 1.0f;               // self-loop
}

__global__ void k_deg_accum(const long long* __restrict__ dst, int E, float* deg) {
    int e = blockIdx.x * blockDim.x + threadIdx.x;
    if (e < E) atomicAdd(&deg[(int)dst[e]], 1.0f);
}

__global__ void k_dinv(float* deg) {
    int i = blockIdx.x * blockDim.x + threadIdx.x;
    if (i < NREG) deg[i] = rsqrtf(deg[i]);     // in-place -> dinv
}

// ------------------------------------------------------------------
// f32 -> bf16 convert
// ------------------------------------------------------------------
__global__ void k_cvt_bf16(const float* __restrict__ src, bf16_t* __restrict__ dst, int n) {
    int i = blockIdx.x * blockDim.x + threadIdx.x;
    if (i < n) dst[i] = to_bf(src[i]);
}

// ------------------------------------------------------------------
// Generic bf16 WMMA GEMM: out = act(A[MxK] @ B[KxNc] + bias)
// grid.x = M/64 (4 waves * 16 rows), grid.y = Nc/64, block = 128
// The shared K x 64 B-panel is staged once per block into LDS by the
// Tensor Data Mover (tensor_load_to_lds, TENSORcnt), then consumed via
// 2-byte LDS reads by all waves.
// ISA layouts (CDNA5 ISA):
//   A 16x32 bf16: lane L -> row m=L%16; elems 0..7 = K k0+(L/16)*8+e,
//                 elems 8..15 = K k0+16+(L/16)*8+(e-8)   (two 16B loads)
//   B 32x16 bf16: lane L -> col nc+L%16; elem e -> same K as A elem e (2-byte LDS reads)
//   D 16x16 f32 : elem r, lane L -> row (L/16)*8+r, col L%16
// ------------------------------------------------------------------
__global__ void k_gemm_bf16(const bf16_t* __restrict__ A, const bf16_t* __restrict__ B,
                            const float* __restrict__ bias,
                            float* __restrict__ outF, bf16_t* __restrict__ outBF,
                            int M, int K, int Nc, int relu) {
    __shared__ bf16_t lds_b[CMID * 64];        // max K=256 -> 32 KB

    const int lane = threadIdx.x & 31;
    const int wave = threadIdx.x >> 5;
    const int mrow = blockIdx.x * 64 + wave * 16;
    const int nc0  = blockIdx.y * 64;
    const int m  = lane & 15;
    const int hh = lane >> 4;

#if __has_builtin(__builtin_amdgcn_tensor_load_to_lds) && __has_builtin(__builtin_amdgcn_s_wait_tensorcnt)
    if (threadIdx.x < 32) {
        // ---- Tensor DMA descriptor (CDNA5 ISA) ----
        const unsigned lds_off = (unsigned)(unsigned long long)(&lds_b[0]);
        const unsigned long long gaddr =
            (unsigned long long)(const void*)(B + (size_t)nc0);   // tile start
        u32x4 g0;
        g0[0] = 1u;                                   // count=1 (valid), no gather
        g0[1] = lds_off;                              // lds_addr
        g0[2] = (unsigned)(gaddr & 0xFFFFFFFFu);      // global_addr[31:0]
        g0[3] = (unsigned)((gaddr >> 32) & 0x01FFFFFFu) | 0x80000000u; // addr[56:32] | type=2
        i32x8 g1;
        g1[0] = 0x00010000;                           // data_size=1 (2 bytes), mask=0
        g1[1] = (int)(((unsigned)Nc & 0xFFFFu) << 16);            // tensor_dim0 lo16
        g1[2] = (int)((((unsigned)Nc >> 16) & 0xFFFFu) |
                      (((unsigned)K & 0xFFFFu) << 16));           // dim0 hi | dim1 lo
        g1[3] = (int)((((unsigned)K >> 16) & 0xFFFFu) | (64u << 16)); // dim1 hi | tile_dim0=64
        g1[4] = (int)((unsigned)K & 0xFFFFu);         // tile_dim1=K, tile_dim2=0
        g1[5] = (int)Nc;                              // tensor_dim0_stride lo32
        g1[6] = 0;                                    // stride0 hi | stride1 lo
        g1[7] = 0;                                    // stride1 hi
        i32x4 z4 = {0, 0, 0, 0};
#if __clang_major__ >= 23
        i32x8 z8 = {0, 0, 0, 0, 0, 0, 0, 0};
        __builtin_amdgcn_tensor_load_to_lds(g0, g1, z4, z4, z8, 0);
#else
        __builtin_amdgcn_tensor_load_to_lds(g0, g1, z4, z4, 0);
#endif
        __builtin_amdgcn_s_wait_tensorcnt(0);
    }
    __syncthreads();
#else
    // fallback: cooperative staging via plain loads
    for (int idx = threadIdx.x; idx < K * 8; idx += blockDim.x) {
        int k = idx >> 3, t = idx & 7;
        *(bf16x8*)&lds_b[k * 64 + t * 8] = *(const bf16x8*)&B[(size_t)k * Nc + nc0 + t * 8];
    }
    __syncthreads();
#endif

    f32x8 acc[4] = {};

    for (int k0 = 0; k0 < K; k0 += 32) {
        BF16V a;
        const bf16_t* ap = A + (size_t)(mrow + m) * K + k0 + hh * 8;
        a.h[0] = *(const bf16x8*)(ap);
        a.h[1] = *(const bf16x8*)(ap + 16);

        const bf16_t* bp = lds_b + (k0 + hh * 8) * 64 + m;   // ds_load path
#pragma unroll
        for (int t = 0; t < 4; ++t) {
            BF16V b;
#pragma unroll
            for (int s = 0; s < 8; ++s) {
                b.h[0][s] = bp[s * 64 + t * 16];
                b.h[1][s] = bp[(16 + s) * 64 + t * 16];
            }
            acc[t] = __builtin_amdgcn_wmma_f32_16x16x32_bf16(
                false, a.v, false, b.v, (short)0, acc[t], false, false);
        }
    }

#pragma unroll
    for (int t = 0; t < 4; ++t) {
        const int col = nc0 + t * 16 + m;
        const float bv = bias ? bias[col] : 0.0f;
#pragma unroll
        for (int r = 0; r < 8; ++r) {
            const int row = mrow + hh * 8 + r;
            float v = acc[t][r] + bv;
            if (relu) v = fmaxf(v, 0.0f);
            const size_t o = (size_t)row * Nc + col;
            if (outF)  outF[o]  = v;
            if (outBF) outBF[o] = to_bf(v);
        }
    }
}

// ------------------------------------------------------------------
// self-loop contribution: agg = xw * dinv^2 (initializes accumulators)
// ------------------------------------------------------------------
__global__ void k_selfloop(const float* __restrict__ xw1, const float* __restrict__ xwp,
                           const float* __restrict__ dinv,
                           float* __restrict__ agg1, float* __restrict__ aggp) {
    int idx = blockIdx.x * blockDim.x + threadIdx.x;
    if (idx >= NREG * CIN) return;
    int i = idx >> 7;
    float s = dinv[i] * dinv[i];
    agg1[idx] = xw1[idx] * s;
    aggp[idx] = xwp[idx] * s;
}

// ------------------------------------------------------------------
// edge scatter: one wave per edge, float4 per lane, f32 global atomics
// ------------------------------------------------------------------
__global__ void k_scatter(const long long* __restrict__ src, const long long* __restrict__ dst,
                          const float* __restrict__ dinv,
                          const float* __restrict__ xw1, const float* __restrict__ xwp,
                          float* __restrict__ agg1, float* __restrict__ aggp, int E) {
    int e = blockIdx.x * 8 + (threadIdx.x >> 5);
    if (e >= E) return;
    const int lane = threadIdx.x & 31;
    const int s = (int)src[e];
    const int d = (int)dst[e];
    const float nrm = dinv[s] * dinv[d];

    const float4* xs1 = (const float4*)(xw1 + (size_t)s * CIN);
    const float4* xsp = (const float4*)(xwp + (size_t)s * CIN);
    __builtin_prefetch(xsp + lane, 0, 0);
    float4 v1 = xs1[lane];
    float4 vp = xsp[lane];
    float* a1 = agg1 + (size_t)d * CIN + lane * 4;
    float* ap = aggp + (size_t)d * CIN + lane * 4;
    atomicAdd(a1 + 0, v1.x * nrm);
    atomicAdd(a1 + 1, v1.y * nrm);
    atomicAdd(a1 + 2, v1.z * nrm);
    atomicAdd(a1 + 3, v1.w * nrm);
    atomicAdd(ap + 0, vp.x * nrm);
    atomicAdd(ap + 1, vp.y * nrm);
    atomicAdd(ap + 2, vp.z * nrm);
    atomicAdd(ap + 3, vp.w * nrm);
}

// ------------------------------------------------------------------
// post-conv: h = relu(agg1 + b1) + x  (-> bf16 for lin1)
//            prep = relu(aggp + bp) + x (-> f32 for lin1p)
// ------------------------------------------------------------------
__global__ void k_post(const float* __restrict__ agg1, const float* __restrict__ aggp,
                       const float* __restrict__ x,
                       const float* __restrict__ b1, const float* __restrict__ bp,
                       bf16_t* __restrict__ hbf, float* __restrict__ prep) {
    int idx = blockIdx.x * blockDim.x + threadIdx.x;
    if (idx >= NREG * CIN) return;
    int c = idx & (CIN - 1);
    float xv = x[idx];
    float h  = fmaxf(agg1[idx] + b1[c], 0.0f) + xv;
    float pp = fmaxf(aggp[idx] + bp[c], 0.0f) + xv;
    hbf[idx]  = to_bf(h);
    prep[idx] = pp;
}

// ------------------------------------------------------------------
// lin1p: p[i,j] = relu(dot(prep[i,:], W[:,j]) + b[j]), PM=8
// ------------------------------------------------------------------
__global__ void k_lin1p(const float* __restrict__ prep, const float* __restrict__ W,
                        const float* __restrict__ b, float* __restrict__ p) {
    int t = blockIdx.x * blockDim.x + threadIdx.x;
    if (t >= NREG * PM) return;
    int i = t >> 3, j = t & 7;
    const float* row = prep + (size_t)i * CIN;
    float acc = b[j];
    for (int c = 0; c < CIN; ++c) acc += row[c] * W[c * PM + j];
    p[t] = fmaxf(acc, 0.0f);
}

// pw[i,q] = sum_p p[i,p] * bilin[p,q]
__global__ void k_bilin(const float* __restrict__ p, const float* __restrict__ bw,
                        float* __restrict__ pw) {
    int t = blockIdx.x * blockDim.x + threadIdx.x;
    if (t >= NREG * PM) return;
    int i = t >> 3, q = t & 7;
    const float* row = p + (size_t)i * PM;
    float acc = 0.0f;
#pragma unroll
    for (int j = 0; j < PM; ++j) acc += row[j] * bw[j * PM + q];
    pw[t] = acc;
}

// acc[i] = dot(h2f[i,:], lin3_W) + lin3_b
__global__ void k_lin3(const float* __restrict__ h2f, const float* __restrict__ W,
                       const float* __restrict__ b, float* __restrict__ out) {
    int i = blockIdx.x * blockDim.x + threadIdx.x;
    if (i >= NREG) return;
    const float* row = h2f + (size_t)i * CMID;
    float acc = b[0];
    for (int c = 0; c < CMID; ++c) acc += row[c] * W[c];
    out[i] = acc;
}

// ------------------------------------------------------------------
// price[i,j] = dot(pw[i,:], p[j,:]) + prices[0]  via V_WMMA_F32_16X16X4_F32
// grid = (N/16, N/128), block = 256 (8 waves); wave -> one 16x16 tile
// A 16x4 f32: lane L -> row L%16; elem v -> k=(L/16)*2+v
// B 4x16 f32: lane L -> col L%16; elem v -> k=(L/16)*2+v  (B = p^T)
// ------------------------------------------------------------------
__global__ void k_price(const float* __restrict__ pw, const float* __restrict__ p,
                        const float* __restrict__ prices, float* __restrict__ out) {
    const int lane = threadIdx.x & 31;
    const int wave = threadIdx.x >> 5;
    const int mt = blockIdx.x;
    const int nt = blockIdx.y * 8 + wave;
    const int m  = lane & 15;
    const int hh = lane >> 4;

    const float* ap = pw + (size_t)(mt * 16 + m) * PM + hh * 2;
    const float* bp = p  + (size_t)(nt * 16 + m) * PM + hh * 2;

    f32x8 acc = {};
#pragma unroll
    for (int k0 = 0; k0 < PM; k0 += 4) {
        f32x2 a = *(const f32x2*)(ap + k0);
        f32x2 b = *(const f32x2*)(bp + k0);
        acc = __builtin_amdgcn_wmma_f32_16x16x4_f32(
            false, a, false, b, (short)0, acc, false, false);
    }

    const float pr = prices[0];
    const size_t col = (size_t)nt * 16 + m;
#pragma unroll
    for (int r = 0; r < 8; ++r) {
        const size_t row = (size_t)mt * 16 + hh * 8 + r;
        out[row * NREG + col] = acc[r] + pr;
    }
}

// ------------------------------------------------------------------
// workspace layout (bytes)
// ------------------------------------------------------------------
#define SZ_NF   ((size_t)NREG * 4)                 // N f32
#define SZ_NC   ((size_t)NREG * CIN * 4)           // N*128 f32
#define SZ_NM   ((size_t)NREG * CMID * 4)          // N*256 f32
#define SZ_NP   ((size_t)NREG * PM * 4)            // N*8 f32
#define SZ_NCB  ((size_t)NREG * CIN * 2)           // N*128 bf16
#define SZ_NMB  ((size_t)NREG * CMID * 2)          // N*256 bf16

extern "C" void kernel_launch(void* const* d_in, const int* in_sizes, int n_in,
                              void* d_out, int out_size, void* d_ws, size_t ws_size,
                              hipStream_t stream) {
    const float*     x       = (const float*)d_in[0];
    const long long* eidx    = (const long long*)d_in[1];   // int64 in reference
    const float*     conv1_W = (const float*)d_in[2];
    const float*     conv1_b = (const float*)d_in[3];
    const float*     convp_W = (const float*)d_in[4];
    const float*     convp_b = (const float*)d_in[5];
    const float*     lin1_W  = (const float*)d_in[6];
    const float*     lin1_b  = (const float*)d_in[7];
    const float*     lin2_W  = (const float*)d_in[8];
    const float*     lin2_b  = (const float*)d_in[9];
    const float*     lin3_W  = (const float*)d_in[10];
    const float*     lin3_b  = (const float*)d_in[11];
    const float*     lin1p_W = (const float*)d_in[12];
    const float*     lin1p_b = (const float*)d_in[13];
    const float*     bilin_W = (const float*)d_in[14];
    const float*     prices  = (const float*)d_in[15];

    const int E = in_sizes[1] / 2;
    const long long* esrc = eidx;
    const long long* edst = eidx + E;

    char* ws = (char*)d_ws;
    size_t off = 0;
    float*  deg  = (float*)(ws + off); off += SZ_NF;      // deg -> dinv in place
    float*  xw1  = (float*)(ws + off); off += SZ_NC;
    float*  xwp  = (float*)(ws + off); off += SZ_NC;
    float*  agg1 = (float*)(ws + off); off += SZ_NC;
    float*  aggp = (float*)(ws + off); off += SZ_NC;
    float*  prep = (float*)(ws + off); off += SZ_NC;
    float*  h2f  = (float*)(ws + off); off += SZ_NM;
    float*  pbuf = (float*)(ws + off); off += SZ_NP;
    float*  pwb  = (float*)(ws + off); off += SZ_NP;
    bf16_t* xbf  = (bf16_t*)(ws + off); off += SZ_NCB;
    bf16_t* hbf  = (bf16_t*)(ws + off); off += SZ_NCB;
    bf16_t* h1bf = (bf16_t*)(ws + off); off += SZ_NMB;
    bf16_t* w1bf = (bf16_t*)(ws + off); off += (size_t)CIN * CIN * 2;
    bf16_t* wpbf = (bf16_t*)(ws + off); off += (size_t)CIN * CIN * 2;
    bf16_t* l1bf = (bf16_t*)(ws + off); off += (size_t)CIN * CMID * 2;
    bf16_t* l2bf = (bf16_t*)(ws + off); off += (size_t)CMID * CMID * 2;

    float* out_acc   = (float*)d_out;
    float* out_price = (float*)d_out + NREG;

    // --- degree / dinv ---
    k_deg_init<<<NREG / 256, 256, 0, stream>>>(deg);
    k_deg_accum<<<(E + 255) / 256, 256, 0, stream>>>(edst, E, deg);
    k_dinv<<<NREG / 256, 256, 0, stream>>>(deg);

    // --- bf16 converts ---
    k_cvt_bf16<<<(NREG * CIN + 255) / 256, 256, 0, stream>>>(x, xbf, NREG * CIN);
    k_cvt_bf16<<<(CIN * CIN + 255) / 256, 256, 0, stream>>>(conv1_W, w1bf, CIN * CIN);
    k_cvt_bf16<<<(CIN * CIN + 255) / 256, 256, 0, stream>>>(convp_W, wpbf, CIN * CIN);
    k_cvt_bf16<<<(CIN * CMID + 255) / 256, 256, 0, stream>>>(lin1_W, l1bf, CIN * CMID);
    k_cvt_bf16<<<(CMID * CMID + 255) / 256, 256, 0, stream>>>(lin2_W, l2bf, CMID * CMID);

    // --- xw = x @ W (both convs), bf16 WMMA, f32 out ---
    {
        dim3 g(NREG / 64, CIN / 64);
        k_gemm_bf16<<<g, 128, 0, stream>>>(xbf, w1bf, nullptr, xw1, nullptr, NREG, CIN, CIN, 0);
        k_gemm_bf16<<<g, 128, 0, stream>>>(xbf, wpbf, nullptr, xwp, nullptr, NREG, CIN, CIN, 0);
    }

    // --- aggregation: self loops then edge scatter ---
    k_selfloop<<<NREG * CIN / 256, 256, 0, stream>>>(xw1, xwp, deg, agg1, aggp);
    k_scatter<<<(E + 7) / 8, 256, 0, stream>>>(esrc, edst, deg, xw1, xwp, agg1, aggp, E);

    // --- relu + bias + residual ---
    k_post<<<NREG * CIN / 256, 256, 0, stream>>>(agg1, aggp, x, conv1_b, convp_b, hbf, prep);

    // --- MLP head: lin1 (relu, bf16 out), lin2 (relu, f32 out), lin3 ---
    {
        dim3 g1(NREG / 64, CMID / 64);
        k_gemm_bf16<<<g1, 128, 0, stream>>>(hbf, l1bf, lin1_b, nullptr, h1bf, NREG, CIN, CMID, 1);
        dim3 g2(NREG / 64, CMID / 64);
        k_gemm_bf16<<<g2, 128, 0, stream>>>(h1bf, l2bf, lin2_b, h2f, nullptr, NREG, CMID, CMID, 1);
    }
    k_lin3<<<NREG / 256, 256, 0, stream>>>(h2f, lin3_W, lin3_b, out_acc);

    // --- price branch ---
    k_lin1p<<<NREG * PM / 256, 256, 0, stream>>>(prep, lin1p_W, lin1p_b, pbuf);
    k_bilin<<<NREG * PM / 256, 256, 0, stream>>>(pbuf, bilin_W, pwb);
    {
        dim3 g(NREG / 16, NREG / 128);
        k_price<<<g, 256, 0, stream>>>(pwb, pbuf, prices, out_price);
    }
}
